// Attention_Rel_Scl_80968723464234
// MI455X (gfx1250) — matrix operand order repaired and checked by reading, not compiled, not run on hardware
//
#include <hip/hip_runtime.h>
#include <hip/hip_bf16.h>

typedef _Float16 v8h  __attribute__((ext_vector_type(8)));
typedef _Float16 v16h __attribute__((ext_vector_type(16)));
typedef float    v8f  __attribute__((ext_vector_type(8)));

union F16x16 { v16h v; v8h h[2]; };

#define B_  8
#define S_  1024
#define E_  512
#define H_  8
#define D_  64
#define NE3 1536   // 3*E

static __device__ __forceinline__ v8f wmma16(const F16x16& a, const F16x16& b, v8f c) {
  return __builtin_amdgcn_wmma_f32_16x16x32_f16(false, a.v, false, b.v, (short)0, c, false, false);
}

// async copy of 16B from global to LDS, tracked by ASYNCcnt
static __device__ __forceinline__ void async_b128(unsigned lds_off, const void* gaddr) {
  asm volatile("global_load_async_to_lds_b128 %0, %1, off"
               :: "v"(lds_off), "v"((unsigned long long)gaddr) : "memory");
}

// ---------------- prep: f32 -> f16 ----------------
__global__ void cvt_x_kernel(const float* __restrict__ x, _Float16* __restrict__ x16, int n) {
  int i = blockIdx.x * blockDim.x + threadIdx.x;
  if (i < n) x16[i] = (_Float16)x[i];
}

// wT[n*E + k] = w[k*3E + n]
__global__ void cvt_wT_kernel(const float* __restrict__ w, _Float16* __restrict__ wT) {
  int i = blockIdx.x * blockDim.x + threadIdx.x;
  if (i < NE3 * E_) {
    int n = i / E_, k = i % E_;
    wT[i] = (_Float16)w[(size_t)k * NE3 + n];
  }
}

// ---------------- QKV projection GEMM ----------------
// M=8192, N=1536, K=512. One wave -> 32(M) x 64(N): 8 WMMAs per k-step, A reused.
__global__ __launch_bounds__(128) void qkv_gemm_kernel(
    const _Float16* __restrict__ x16, const _Float16* __restrict__ wT,
    _Float16* __restrict__ q16, _Float16* __restrict__ k16, _Float16* __restrict__ vT16) {
  int wave = (blockIdx.x * blockDim.x + threadIdx.x) >> 5;
  int lane = threadIdx.x & 31;
  int lm = lane & 15, lh = lane >> 4;

  int nst = wave % (NE3 / 64);          // 0..23
  int mt  = wave / (NE3 / 64);          // 0..255
  int m0 = mt * 32, n0 = nst * 64;

  v8f acc[2][4] = {};
  const _Float16* arow0 = x16 + (size_t)(m0 + lm) * E_ + lh * 8;
  const _Float16* arow1 = arow0 + (size_t)16 * E_;

  for (int k0 = 0; k0 < E_; k0 += 32) {
    F16x16 a0, a1;
    a0.h[0] = *(const v8h*)(arow0 + k0);
    a0.h[1] = *(const v8h*)(arow0 + k0 + 16);
    a1.h[0] = *(const v8h*)(arow1 + k0);
    a1.h[1] = *(const v8h*)(arow1 + k0 + 16);
    F16x16 b[4];
#pragma unroll
    for (int t = 0; t < 4; ++t) {
      const _Float16* brow = wT + (size_t)(n0 + t * 16 + lm) * E_ + k0 + lh * 8;
      b[t].h[0] = *(const v8h*)(brow);
      b[t].h[1] = *(const v8h*)(brow + 16);
    }
#pragma unroll
    for (int t = 0; t < 4; ++t) {
      acc[0][t] = wmma16(a0, b[t], acc[0][t]);
      acc[1][t] = wmma16(a1, b[t], acc[1][t]);
    }
  }

  int sec = nst >> 3;                   // 0=q 1=k 2=v
  int h   = nst & 7;
  float scale = (sec == 0) ? 0.125f : 1.0f;  // D^-0.5 folded into Q
#pragma unroll
  for (int mi = 0; mi < 2; ++mi) {
#pragma unroll
    for (int t = 0; t < 4; ++t) {
#pragma unroll
      for (int r = 0; r < 8; ++r) {
        int m = m0 + mi * 16 + r + 8 * lh;
        int bb = m >> 10, s = m & (S_ - 1);
        int d = t * 16 + lm;
        _Float16 val = (_Float16)(acc[mi][t][r] * scale);
        if (sec == 0)
          q16[(((size_t)bb * H_ + h) * S_ + s) * D_ + d] = val;
        else if (sec == 1)
          k16[(((size_t)bb * H_ + h) * S_ + s) * D_ + d] = val;
        else
          vT16[(((size_t)bb * H_ + h) * D_ + d) * S_ + s] = val;
      }
    }
  }
}

// ---------------- flash attention with relative position bias ----------------
// grid: (B*H, S/64); block 128 = 4 waves; each wave owns 16 q rows.
// K/V chunks staged in LDS via async global->LDS, double buffered, shared by 4 waves.
__global__ __launch_bounds__(128) void attn_kernel(
    const _Float16* __restrict__ q16, const _Float16* __restrict__ k16,
    const _Float16* __restrict__ vT16, const float* __restrict__ bias_table,
    float* __restrict__ out) {
  int bh = blockIdx.x;                  // 0..63
  int b  = bh >> 3, h = bh & 7;
  int tid = threadIdx.x;
  int wid = tid >> 5, lane = tid & 31;
  int lm = lane & 15, lh = lane >> 4;
  int qbase = blockIdx.y * 64 + wid * 16;

  __shared__ __align__(16) _Float16 s_k[2][32][64];   // K chunk  [kpos][d]
  __shared__ __align__(16) _Float16 s_v[2][64][32];   // V^T chunk [d][kpos]
  __shared__ __align__(16) float    s_sc[4][16][32];
  __shared__ __align__(16) _Float16 s_p [4][16][32];
  __shared__ __align__(16) float    s_row[4][16];

  const _Float16* Kb = k16  + (size_t)bh * S_ * D_;
  const _Float16* Vb = vT16 + (size_t)bh * D_ * S_;

  // Q fragments (rows qbase..+15, d split into two K=32 steps)
  const _Float16* qrow = q16 + ((size_t)bh * S_ + qbase + lm) * D_ + lh * 8;
  F16x16 qa[2];
  qa[0].h[0] = *(const v8h*)(qrow + 0);
  qa[0].h[1] = *(const v8h*)(qrow + 16);
  qa[1].h[0] = *(const v8h*)(qrow + 32);
  qa[1].h[1] = *(const v8h*)(qrow + 48);

  // issue async loads for one 32-wide k-chunk into buffer `buf`
  auto issue_chunk = [&](int buf, int k0) {
    unsigned kdst = (unsigned)(uintptr_t)&s_k[buf][0][0];
    const _Float16* ksrc = Kb + (size_t)k0 * D_;      // 4KB contiguous
#pragma unroll
    for (int it = 0; it < 2; ++it) {
      int j = tid + it * 128;                          // 0..255 transfers of 16B
      async_b128(kdst + j * 16, ksrc + j * 8);
    }
    unsigned vdst = (unsigned)(uintptr_t)&s_v[buf][0][0];
#pragma unroll
    for (int it = 0; it < 2; ++it) {
      int j = tid + it * 128;
      int row = j >> 2, cc = j & 3;                    // 64 rows x 4 chunks of 16B
      async_b128(vdst + (unsigned)(row * 64 + cc * 16),
                 Vb + (size_t)row * S_ + k0 + cc * 8);
    }
  };

  v8f o[4] = {};
  float rm = -1e30f, rl = 0.f;

  issue_chunk(0, 0);

  for (int ci = 0; ci < S_ / 32; ++ci) {
    int k0 = ci * 32;
    int buf = ci & 1;

    if (ci + 1 < S_ / 32) {
      issue_chunk(buf ^ 1, k0 + 32);
      asm volatile("s_wait_asynccnt 0x4" ::: "memory");  // chunk ci's 4 ops done
    } else {
      asm volatile("s_wait_asynccnt 0x0" ::: "memory");
    }
    __syncthreads();                                     // peers' loads landed

    // QK^T: 2 col-tiles x 2 d-steps, B fragments from LDS
    v8f sc[2] = {};
#pragma unroll
    for (int t = 0; t < 2; ++t) {
#pragma unroll
      for (int s = 0; s < 2; ++s) {
        const _Float16* krow = &s_k[buf][t * 16 + lm][s * 32 + lh * 8];
        F16x16 bf;
        bf.h[0] = *(const v8h*)(krow);
        bf.h[1] = *(const v8h*)(krow + 16);
        sc[t] = wmma16(qa[s], bf, sc[t]);
      }
    }

    // + rel bias, spill scores to wave-private LDS
#pragma unroll
    for (int t = 0; t < 2; ++t) {
#pragma unroll
      for (int r = 0; r < 8; ++r) {
        int q = qbase + r + 8 * lh;
        int k = k0 + t * 16 + lm;
        float v = sc[t][r] + bias_table[(size_t)(q - k + (S_ - 1)) * H_ + h];
        s_sc[wid][r + 8 * lh][t * 16 + lm] = v;
      }
    }
    asm volatile("s_wait_dscnt 0x0" ::: "memory");

    // online softmax: lane owns row lm (duplicated in both halves)
    float vals[32];
    float cmax = -1e30f;
#pragma unroll
    for (int j = 0; j < 32; ++j) { vals[j] = s_sc[wid][lm][j]; cmax = fmaxf(cmax, vals[j]); }
    float mnew  = fmaxf(rm, cmax);
    float alpha = __expf(rm - mnew);
    float csum  = 0.f;
#pragma unroll
    for (int j = 0; j < 32; ++j) {
      float p = __expf(vals[j] - mnew);
      csum += p;
      s_p[wid][lm][j] = (_Float16)p;
    }
    rl = rl * alpha + csum;
    rm = mnew;
    s_row[wid][lm] = alpha;
    asm volatile("s_wait_dscnt 0x0" ::: "memory");

    // rescale accumulated O by per-row alpha
#pragma unroll
    for (int r = 0; r < 8; ++r) {
      float a = s_row[wid][r + 8 * lh];
      o[0][r] *= a; o[1][r] *= a; o[2][r] *= a; o[3][r] *= a;
    }

    // reload P in A-fragment layout from LDS
    F16x16 pa;
    const _Float16* pr = &s_p[wid][lm][lh * 8];
    pa.h[0] = *(const v8h*)(pr);
    pa.h[1] = *(const v8h*)(pr + 16);

    // P x V: 4 d-tiles, B fragments from LDS
#pragma unroll
    for (int dt = 0; dt < 4; ++dt) {
      const _Float16* vrow = &s_v[buf][dt * 16 + lm][lh * 8];
      F16x16 bf;
      bf.h[0] = *(const v8h*)(vrow);
      bf.h[1] = *(const v8h*)(vrow + 16);
      o[dt] = wmma16(pa, bf, o[dt]);
    }

    __syncthreads();                                     // buffer reuse safety
  }

  // finalize: divide by row sum, store [B,S,H,D]
  s_row[wid][lm] = rl;
  asm volatile("s_wait_dscnt 0x0" ::: "memory");
#pragma unroll
  for (int r = 0; r < 8; ++r) {
    int q = qbase + r + 8 * lh;
    float inv = 1.0f / s_row[wid][r + 8 * lh];
    float* orow = out + ((size_t)b * S_ + q) * E_ + h * D_;
#pragma unroll
    for (int dt = 0; dt < 4; ++dt) orow[dt * 16 + lm] = o[dt][r] * inv;
  }
}

// ---------------- LayerNorm over E=512, wave per row ----------------
__global__ __launch_bounds__(256) void ln_kernel(
    const float* __restrict__ in, const float* __restrict__ gamma,
    const float* __restrict__ beta, float* __restrict__ out) {
  int row  = blockIdx.x * (blockDim.x >> 5) + (threadIdx.x >> 5);
  int lane = threadIdx.x & 31;
  const float* p = in + (size_t)row * E_;
  float4 v[4];
  float sum = 0.f, sq = 0.f;
#pragma unroll
  for (int i = 0; i < 4; ++i) {
    v[i] = *(const float4*)(p + lane * 4 + i * 128);
    sum += v[i].x + v[i].y + v[i].z + v[i].w;
    sq  += v[i].x * v[i].x + v[i].y * v[i].y + v[i].z * v[i].z + v[i].w * v[i].w;
  }
#pragma unroll
  for (int off = 16; off > 0; off >>= 1) {
    sum += __shfl_xor(sum, off);
    sq  += __shfl_xor(sq,  off);
  }
  float mean = sum * (1.0f / E_);
  float var  = sq * (1.0f / E_) - mean * mean;
  float rstd = rsqrtf(var + 1e-5f);
  float* q = out + (size_t)row * E_;
#pragma unroll
  for (int i = 0; i < 4; ++i) {
    int c = lane * 4 + i * 128;
    float4 g = *(const float4*)(gamma + c);
    float4 bb = *(const float4*)(beta + c);
    float4 r;
    r.x = (v[i].x - mean) * rstd * g.x + bb.x;
    r.y = (v[i].y - mean) * rstd * g.y + bb.y;
    r.z = (v[i].z - mean) * rstd * g.z + bb.z;
    r.w = (v[i].w - mean) * rstd * g.w + bb.w;
    *(float4*)(q + c) = r;
  }
}

extern "C" void kernel_launch(void* const* d_in, const int* in_sizes, int n_in,
                              void* d_out, int out_size, void* d_ws, size_t ws_size,
                              hipStream_t stream) {
  const float* x     = (const float*)d_in[0];
  const float* w_qkv = (const float*)d_in[1];
  const float* bias  = (const float*)d_in[2];
  const float* gamma = (const float*)d_in[3];
  const float* beta  = (const float*)d_in[4];
  float* out = (float*)d_out;

  const size_t ME   = (size_t)B_ * S_ * E_;   // 4,194,304
  const size_t BHSD = (size_t)B_ * H_ * S_ * D_;

  char* ws = (char*)d_ws;
  _Float16* x16  = (_Float16*)ws;                 ws += ME * 2;
  _Float16* wT   = (_Float16*)ws;                 ws += (size_t)NE3 * E_ * 2;
  _Float16* q16  = (_Float16*)ws;                 ws += BHSD * 2;
  _Float16* k16  = (_Float16*)ws;                 ws += BHSD * 2;
  _Float16* vT16 = (_Float16*)ws;                 ws += BHSD * 2;
  float*    attno = (float*)ws;                   // B*S*E f32

  cvt_x_kernel<<<(int)(ME / 256), 256, 0, stream>>>(x, x16, (int)ME);
  cvt_wT_kernel<<<(NE3 * E_) / 256, 256, 0, stream>>>(w_qkv, wT);

  // 256 m-tiles * 24 n-supertiles = 6144 waves, 4 waves/block
  qkv_gemm_kernel<<<6144 / 4, 128, 0, stream>>>(x16, wT, q16, k16, vT16);

  dim3 agrid(B_ * H_, S_ / 64);
  attn_kernel<<<agrid, 128, 0, stream>>>(q16, k16, vT16, bias, attno);

  ln_kernel<<<(B_ * S_) / 8, 256, 0, stream>>>(attno, gamma, beta, out);
}
